// VoxelGNNGenerator_61100204753234
// MI455X (gfx1250) — matrix operand order repaired and checked by reading, not compiled, not run on hardware
//
#include <hip/hip_runtime.h>

// ---------------------------------------------------------------------------
// Problem constants (match reference)
// ---------------------------------------------------------------------------
#define NV_ 30000           // voxels
#define NL_ 1500            // locals
#define NE_ 180000          // edges
#define NT_TYPES 180000     // T
#define MP_ 30016           // NV_ padded to multiple of 32 (GEMM row tiling)
#define EPS_ 1e-5f

typedef _Float16 half_t;
typedef __attribute__((ext_vector_type(16))) _Float16 v16h;
typedef __attribute__((ext_vector_type(8)))  _Float16 v8h;
typedef __attribute__((ext_vector_type(8)))  float    v8f;

// ---------------------------------------------------------------------------
// Phase 1: type-wise scatter mean
// ---------------------------------------------------------------------------
__global__ void k_scatter_local(const float* __restrict__ lx, const int* __restrict__ lt,
                                float* __restrict__ sums, float* __restrict__ cnt) {
  int i = blockIdx.x * blockDim.x + threadIdx.x;
  if (i >= NL_ * 32) return;
  int node = i >> 5, f = i & 31;
  int t = lt[node];
  atomicAdd(&sums[(size_t)t * 32 + f], lx[i]);
  if (f == 0) atomicAdd(&cnt[t], 1.0f);
}

__global__ void k_gather_matched(const float* __restrict__ sums, const float* __restrict__ cnt,
                                 const int* __restrict__ vt, half_t* __restrict__ out) {
  int i = blockIdx.x * blockDim.x + threadIdx.x;
  if (i >= MP_ * 32) return;
  int v = i >> 5, f = i & 31;
  float val = 0.0f;
  if (v < NV_) {
    int t = vt[v];
    float c = cnt[t];
    if (c > 0.0f) val = sums[(size_t)t * 32 + f] / fmaxf(c, 1.0f);
  }
  out[i] = (half_t)val;
}

// W [K][N] fp32 row-major -> Wt [Npad][K] fp16 (column-major, zero padded cols)
__global__ void k_transpose_w(const float* __restrict__ W, half_t* __restrict__ Wt,
                              int K, int N, int Npad) {
  int i = blockIdx.x * blockDim.x + threadIdx.x;
  if (i >= K * Npad) return;
  int n = i / K, k = i - n * K;
  float v = (n < N) ? W[(size_t)k * N + n] : 0.0f;
  Wt[(size_t)n * K + k] = (half_t)v;
}

// ---------------------------------------------------------------------------
// WMMA fragment loads, per CDNA5 ISA 16-bit layouts (wave32).
// A 16x32 (MxK): lane m=lane&15, g=lane>>4; element e -> K = 16*(e/8)+8g+(e%8)
// B 32x16 (KxN): lane n=lane&15, g=lane>>4; element e -> K = 16*(e/8)+8g+(e%8)
// ---------------------------------------------------------------------------
__device__ __forceinline__ v16h frag_a(const half_t* __restrict__ p, int g) {
  const v8h* p0 = (const v8h*)(p + 8 * g);
  const v8h* p1 = (const v8h*)(p + 16 + 8 * g);
  v8h lo = *p0, hi = *p1;
  v16h a;
#pragma unroll
  for (int i = 0; i < 8; ++i) { a[i] = lo[i]; a[i + 8] = hi[i]; }
  return a;
}
__device__ __forceinline__ v16h frag_b(const half_t* __restrict__ p, int g) {
  const v8h* p0 = (const v8h*)(p + 8 * g);
  const v8h* p1 = (const v8h*)(p + 16 + 8 * g);
  v8h lo = *p0, hi = *p1;
  v16h b;
#pragma unroll
  for (int i = 0; i < 8; ++i) { b[i] = lo[i]; b[i + 8] = hi[i]; }
  return b;
}

// ---------------------------------------------------------------------------
// Fused GEMM: out = A[Mp x K](fp16) @ W[K x Npad] (+bias) [+LayerNorm+lrelu]
// Template NT4 = column tiles per wave; wave wq owns tiles [wq*NT4, wq*NT4+NT4)
// (contiguous, uniform => branch-free WMMA inner loop).
// MODE 0: fp32 output (optional bias), row/col store guards.
// MODE 1: bias + LayerNorm(row over true N) + leaky-relu(0.2) -> fp16 output.
// Block: 256 threads = 8 wave32; 32 rows/block; 4 waves share 16 rows.
// ---------------------------------------------------------------------------
template <int NT4, int MODE>
__global__ __launch_bounds__(256)
void k_gemm_wmma(const half_t* __restrict__ A, const half_t* __restrict__ Wt,
                 const float* __restrict__ bias, const float* __restrict__ gamma,
                 const float* __restrict__ beta, void* __restrict__ outv,
                 int M_store, int K, int N, int out_ld, int n_store) {
  __shared__ float ln_s[32][4], ln_q[32][4], ln_mean[32], ln_rstd[32];
  const int tid  = threadIdx.x;
  const int wave = tid >> 5, lane = tid & 31;
  const int wq = wave & 3, rg = wave >> 2;
  const int g = lane >> 4, nl = lane & 15;
  const int rowbase = blockIdx.x * 32 + rg * 16;
  const int tile0 = wq * NT4;

  v8f acc[NT4];
#pragma unroll
  for (int i = 0; i < NT4; ++i)
#pragma unroll
    for (int j = 0; j < 8; ++j) acc[i][j] = 0.0f;

  const half_t* Arow = A + (size_t)(rowbase + nl) * K;
  const half_t* Bcol = Wt + (size_t)(tile0 * 16 + nl) * K;

  for (int k0 = 0; k0 < K; k0 += 32) {
    v16h a = frag_a(Arow + k0, g);
#pragma unroll
    for (int ti = 0; ti < NT4; ++ti) {
      v16h b = frag_b(Bcol + (size_t)ti * 16 * K + k0, g);
      acc[ti] = __builtin_amdgcn_wmma_f32_16x16x32_f16(
          false, a, false, b, (short)0, acc[ti], false, false);
    }
  }

  if (MODE == 0) {
    float* out = (float*)outv;
#pragma unroll
    for (int ti = 0; ti < NT4; ++ti) {
      int col = (tile0 + ti) * 16 + nl;
      if (col < n_store) {
        float bb = bias ? bias[col] : 0.0f;
#pragma unroll
        for (int r = 0; r < 8; ++r) {
          int row = rowbase + r + 8 * g;
          if (row < M_store) out[(size_t)row * out_ld + col] = acc[ti][r] + bb;
        }
      }
    }
    return;
  }

  // MODE 1: bias
#pragma unroll
  for (int ti = 0; ti < NT4; ++ti) {
    float bb = bias[(tile0 + ti) * 16 + nl];
#pragma unroll
    for (int r = 0; r < 8; ++r) acc[ti][r] += bb;
  }
  // per-row partial sums over this wave's column tiles
  float ps[8], qs[8];
#pragma unroll
  for (int r = 0; r < 8; ++r) { ps[r] = 0.0f; qs[r] = 0.0f; }
#pragma unroll
  for (int ti = 0; ti < NT4; ++ti)
#pragma unroll
    for (int r = 0; r < 8; ++r) { float v = acc[ti][r]; ps[r] += v; qs[r] += v * v; }
#pragma unroll
  for (int r = 0; r < 8; ++r) {
    float s = ps[r], q = qs[r];
#pragma unroll
    for (int m = 1; m <= 8; m <<= 1) { s += __shfl_xor(s, m, 32); q += __shfl_xor(q, m, 32); }
    if (nl == 0) { int mb = rg * 16 + r + 8 * g; ln_s[mb][wq] = s; ln_q[mb][wq] = q; }
  }
  __syncthreads();
  if (tid < 32) {
    float s = ln_s[tid][0] + ln_s[tid][1] + ln_s[tid][2] + ln_s[tid][3];
    float q = ln_q[tid][0] + ln_q[tid][1] + ln_q[tid][2] + ln_q[tid][3];
    float mean = s / (float)N;
    float var  = q / (float)N - mean * mean;
    ln_mean[tid] = mean;
    ln_rstd[tid] = rsqrtf(var + EPS_);
  }
  __syncthreads();
  half_t* out = (half_t*)outv;
#pragma unroll
  for (int ti = 0; ti < NT4; ++ti) {
    int col = (tile0 + ti) * 16 + nl;
    float gm = gamma[col], bt = beta[col];
#pragma unroll
    for (int r = 0; r < 8; ++r) {
      int mb = rg * 16 + r + 8 * g;
      float v = (acc[ti][r] - ln_mean[mb]) * ln_rstd[mb] * gm + bt;
      v = v > 0.0f ? v : 0.2f * v;
      out[(size_t)(rowbase + r + 8 * g) * out_ld + col] = (half_t)v;
    }
  }
}

// ---------------------------------------------------------------------------
// Concat packers (2-D grid: y = row, x covers columns)
// ---------------------------------------------------------------------------
__global__ void k_pack448(const half_t* __restrict__ enc, const float* __restrict__ vx,
                          const float* __restrict__ zf, half_t* __restrict__ out) {
  int f = blockIdx.x * blockDim.x + threadIdx.x;
  int v = blockIdx.y;
  if (f >= 448) return;
  float val = 0.0f;
  if (v < NV_) {
    if (f < 256)      val = (float)enc[(size_t)v * 256 + f];
    else if (f < 320) val = vx[(size_t)v * 64 + (f - 256)];
    else              val = zf[(size_t)v * 128 + (f - 320)];
  }
  out[(size_t)v * 448 + f] = (half_t)val;
}

__global__ void k_pack1472(const half_t* __restrict__ e, const half_t* __restrict__ x,
                           const half_t* __restrict__ enc, const float* __restrict__ vx,
                           const float* __restrict__ zf, half_t* __restrict__ out) {
  int f = blockIdx.x * blockDim.x + threadIdx.x;
  int v = blockIdx.y;
  if (f >= 1472) return;
  float val = 0.0f;
  if (v < NV_) {
    if (f < 512)       val = (float)e[(size_t)v * 512 + f];
    else if (f < 1024) val = (float)x[(size_t)v * 512 + (f - 512)];
    else if (f < 1280) val = (float)enc[(size_t)v * 256 + (f - 1024)];
    else if (f < 1344) val = vx[(size_t)v * 64 + (f - 1280)];
    else               val = zf[(size_t)v * 128 + (f - 1344)];
  }
  out[(size_t)v * 1472 + f] = (half_t)val;
}

// ---------------------------------------------------------------------------
// GCN support kernels
// ---------------------------------------------------------------------------
__global__ void k_edge_deg(const int* __restrict__ dst, float* __restrict__ deg) {
  int e = blockIdx.x * blockDim.x + threadIdx.x;
  if (e >= NE_) return;
  atomicAdd(&deg[dst[e]], 1.0f);
}

__global__ void k_dis(const float* __restrict__ deg, float* __restrict__ dis,
                      float* __restrict__ dis2) {
  int v = blockIdx.x * blockDim.x + threadIdx.x;
  if (v >= MP_) return;
  float d = (v < NV_) ? deg[v] + 1.0f : 1.0f;   // +1 self loop
  dis[v]  = rsqrtf(d);
  dis2[v] = 1.0f / d;
}

// one thread per (edge, 4-feature chunk); nq = N/4 is a power of two (logq)
__global__ void k_edge_agg(const int* __restrict__ src, const int* __restrict__ dst,
                           const float* __restrict__ dis, const float* __restrict__ h,
                           float* __restrict__ agg, int N, int logq) {
  long long i = (long long)blockIdx.x * blockDim.x + threadIdx.x;
  if (i >= ((long long)NE_ << logq)) return;
  int e  = (int)(i >> logq);
  int fc = ((int)i & ((1 << logq) - 1)) << 2;
  int s = src[e], d = dst[e];
  float c = dis[s] * dis[d];
  const float4 hv = *(const float4*)(h + (size_t)s * N + fc);
  float* ap = agg + (size_t)d * N + fc;
  atomicAdd(ap + 0, hv.x * c);
  atomicAdd(ap + 1, hv.y * c);
  atomicAdd(ap + 2, hv.z * c);
  atomicAdd(ap + 3, hv.w * c);
}

// t = agg + h*dis2 + bias   (in-place into h); 2-D grid, y = row
__global__ void k_gcn_combine(float* __restrict__ h, const float* __restrict__ agg,
                              const float* __restrict__ dis2, const float* __restrict__ bias,
                              int N) {
  int f = blockIdx.x * blockDim.x + threadIdx.x;
  int v = blockIdx.y;
  if (f >= N) return;
  size_t i = (size_t)v * N + f;
  h[i] = agg[i] + h[i] * dis2[v] + bias[f];
}

__global__ void k_colsum(const float* __restrict__ t, float* __restrict__ fsum, int N) {
  int v0 = blockIdx.x * 128;
  int v1 = v0 + 128; if (v1 > NV_) v1 = NV_;
  for (int f = threadIdx.x; f < N; f += blockDim.x) {
    float s = 0.0f;
    for (int v = v0; v < v1; ++v) s += t[(size_t)v * N + f];
    atomicAdd(&fsum[f], s);
  }
}

__global__ void k_meanfin(const float* __restrict__ fsum, float* __restrict__ fmean, int N) {
  int f = threadIdx.x;
  if (f < N) fmean[f] = fsum[f] / (float)NV_;
}

__global__ void k_csqsum(const float* __restrict__ t, const float* __restrict__ a,
                         const float* __restrict__ fmean, float* __restrict__ fsq, int N) {
  int v0 = blockIdx.x * 128;
  int v1 = v0 + 128; if (v1 > NV_) v1 = NV_;
  for (int f = threadIdx.x; f < N; f += blockDim.x) {
    float am = a[f] * fmean[f];
    float s = 0.0f;
    for (int v = v0; v < v1; ++v) { float c = t[(size_t)v * N + f] - am; s += c * c; }
    atomicAdd(&fsq[f], s);
  }
}

__global__ void k_varfin(const float* __restrict__ fsq, float* __restrict__ frstd, int N) {
  int f = threadIdx.x;
  if (f < N) frstd[f] = rsqrtf(fsq[f] / (float)NV_ + EPS_);
}

// GraphNorm affine + ReLU -> fp16; 2-D grid, y = row (pad rows zeroed)
__global__ void k_gnorm_out(const float* __restrict__ t, const float* __restrict__ a,
                            const float* __restrict__ fmean, const float* __restrict__ frstd,
                            const float* __restrict__ w, const float* __restrict__ b,
                            half_t* __restrict__ out, int N) {
  int f = blockIdx.x * blockDim.x + threadIdx.x;
  int v = blockIdx.y;
  if (f >= N) return;
  float val = 0.0f;
  if (v < NV_) {
    float c = t[(size_t)v * N + f] - a[f] * fmean[f];
    float o = c * frstd[f] * w[f] + b[f];
    val = o > 0.0f ? o : 0.0f;   // ReLU
  }
  out[(size_t)v * N + f] = (half_t)val;
}

// ---------------------------------------------------------------------------
// Gumbel softmax (tau=1): soft = softmax(logits+g); hard = onehot(argmax(soft))
// ---------------------------------------------------------------------------
__global__ void k_gumbel(const float* __restrict__ logits, const float* __restrict__ gum,
                         float* __restrict__ hard, float* __restrict__ soft) {
  int v = blockIdx.x * blockDim.x + threadIdx.x;
  if (v >= NV_) return;
  float s[8], e[8];
  float mx = -3.0e38f;
#pragma unroll
  for (int j = 0; j < 8; ++j) { s[j] = logits[v * 8 + j] + gum[v * 8 + j]; mx = fmaxf(mx, s[j]); }
  float sum = 0.0f;
#pragma unroll
  for (int j = 0; j < 8; ++j) { e[j] = expf(s[j] - mx); sum += e[j]; }
  float inv = 1.0f / sum;
  int am = 0; float best = s[0];
#pragma unroll
  for (int j = 1; j < 8; ++j) if (s[j] > best) { best = s[j]; am = j; }
#pragma unroll
  for (int j = 0; j < 8; ++j) {
    soft[v * 8 + j] = e[j] * inv;
    hard[v * 8 + j] = (j == am) ? 1.0f : 0.0f;
  }
}

// ---------------------------------------------------------------------------
// Host orchestration
// ---------------------------------------------------------------------------
extern "C" void kernel_launch(void* const* d_in, const int* in_sizes, int n_in,
                              void* d_out, int out_size, void* d_ws, size_t ws_size,
                              hipStream_t stream) {
  (void)in_sizes; (void)n_in; (void)out_size; (void)ws_size;

  const float* local_x    = (const float*)d_in[0];
  const int*   local_type = (const int*)d_in[1];
  const float* voxel_x    = (const float*)d_in[2];
  const int*   voxel_type = (const int*)d_in[3];
  const int*   edge_index = (const int*)d_in[4];
  const float* zf         = (const float*)d_in[5];  // z[0], [N_V, 128]
  const float* gumbel     = (const float*)d_in[6];

  int ix = 7;
  const float *mfe_W[3], *mfe_b[3], *mfe_g[3], *mfe_be[3];
  for (int i = 0; i < 3; ++i) mfe_W[i]  = (const float*)d_in[ix++];
  for (int i = 0; i < 3; ++i) mfe_b[i]  = (const float*)d_in[ix++];
  for (int i = 0; i < 3; ++i) mfe_g[i]  = (const float*)d_in[ix++];
  for (int i = 0; i < 3; ++i) mfe_be[i] = (const float*)d_in[ix++];
  const float *mlp_W[3], *mlp_b[3], *mlp_g[3], *mlp_be[3];
  for (int i = 0; i < 3; ++i) mlp_W[i]  = (const float*)d_in[ix++];
  for (int i = 0; i < 3; ++i) mlp_b[i]  = (const float*)d_in[ix++];
  for (int i = 0; i < 3; ++i) mlp_g[i]  = (const float*)d_in[ix++];
  for (int i = 0; i < 3; ++i) mlp_be[i] = (const float*)d_in[ix++];
  const float *conv_W[6], *conv_b[6], *gn_w[6], *gn_b[6], *gn_a[6];
  for (int i = 0; i < 6; ++i) conv_W[i] = (const float*)d_in[ix++];
  for (int i = 0; i < 6; ++i) conv_b[i] = (const float*)d_in[ix++];
  for (int i = 0; i < 6; ++i) gn_w[i]   = (const float*)d_in[ix++];
  for (int i = 0; i < 6; ++i) gn_b[i]   = (const float*)d_in[ix++];
  for (int i = 0; i < 6; ++i) gn_a[i]   = (const float*)d_in[ix++];
  const float *dec_W[5], *dec_b[5], *dec_g[4], *dec_be[4];
  for (int i = 0; i < 5; ++i) dec_W[i]  = (const float*)d_in[ix++];
  for (int i = 0; i < 5; ++i) dec_b[i]  = (const float*)d_in[ix++];
  for (int i = 0; i < 4; ++i) dec_g[i]  = (const float*)d_in[ix++];
  for (int i = 0; i < 4; ++i) dec_be[i] = (const float*)d_in[ix++];

  // ---- workspace arena (regions aliased across sequential phases) ----
  char* base = (char*)d_ws;
  size_t off = 0;
  auto alloc = [&](size_t bytes) -> char* {
    char* r = base + off;
    off = (off + bytes + 255) & ~(size_t)255;
    return r;
  };
  float* big1 = (float*)alloc((size_t)MP_ * 512 * 4);  // sums | GCN h/t fp32
  float* big2 = (float*)alloc((size_t)MP_ * 512 * 4);  // agg fp32 (contiguous w/ big1)
  float* sums  = big1;
  float* cnt   = big1 + (size_t)NT_TYPES * 32;
  float* hbuff = big1;                    // GCN matmul / combined tensor
  float* agg   = big2;
  half_t* concat = (half_t*)big1;         // decoder concat spans big1+big2

  half_t* matched = (half_t*)alloc((size_t)MP_ * 32 * 2);
  half_t* enc     = (half_t*)alloc((size_t)MP_ * 256 * 2);
  half_t* hb1     = (half_t*)alloc((size_t)MP_ * 512 * 2);
  half_t* hb2     = (half_t*)alloc((size_t)MP_ * 512 * 2);
  half_t* xb      = (half_t*)alloc((size_t)MP_ * 512 * 2);
  float* deg   = (float*)alloc((size_t)MP_ * 4);
  float* dis   = (float*)alloc((size_t)MP_ * 4);
  float* dis2  = (float*)alloc((size_t)MP_ * 4);
  float* fsum  = (float*)alloc(512 * 4);
  float* fsq   = (float*)alloc(512 * 4);
  float* fmean = (float*)alloc(512 * 4);
  float* frstd = (float*)alloc(512 * 4);

  // ---- weight transposition (fp32 [K][N] -> fp16 [Npad][K]) ----
  const int NW = 17;
  const float* Wsrc[NW] = { mfe_W[0], mfe_W[1], mfe_W[2],
                            mlp_W[0], mlp_W[1], mlp_W[2],
                            conv_W[0], conv_W[1], conv_W[2], conv_W[3], conv_W[4], conv_W[5],
                            dec_W[0], dec_W[1], dec_W[2], dec_W[3], dec_W[4] };
  const int Wk[NW] = { 32, 256, 256, 448, 512, 512,
                       512, 256, 128, 64, 128, 256,
                       1472, 512, 256, 128, 64 };
  const int Wn[NW] = { 256, 256, 256, 512, 512, 512,
                       256, 128, 64, 128, 256, 512,
                       512, 256, 128, 64, 8 };
  const int Wp[NW] = { 256, 256, 256, 512, 512, 512,
                       256, 128, 64, 128, 256, 512,
                       512, 256, 128, 64, 64 };   // last padded to 64 cols
  half_t* Wt[NW];
  for (int i = 0; i < NW; ++i) Wt[i] = (half_t*)alloc((size_t)Wk[i] * Wp[i] * 2);
  for (int i = 0; i < NW; ++i) {
    int n = Wk[i] * Wp[i];
    k_transpose_w<<<(n + 255) / 256, 256, 0, stream>>>(Wsrc[i], Wt[i], Wk[i], Wn[i], Wp[i]);
  }

  // GEMM dispatch: Npad = NT4*64; mode 0 = fp32 out, mode 1 = LN+lrelu fp16
  auto gemm = [&](const half_t* A, const half_t* W, const float* bb, const float* gg,
                  const float* be, void* out, int M_store, int K, int Npad, int mode,
                  int N_true, int out_ld, int n_store) {
    dim3 gsz(MP_ / 32);
    int nt4 = Npad >> 6;
    if (mode == 1) {
      switch (nt4) {
        case 1: k_gemm_wmma<1,1><<<gsz,256,0,stream>>>(A,W,bb,gg,be,out,M_store,K,N_true,out_ld,n_store); break;
        case 2: k_gemm_wmma<2,1><<<gsz,256,0,stream>>>(A,W,bb,gg,be,out,M_store,K,N_true,out_ld,n_store); break;
        case 4: k_gemm_wmma<4,1><<<gsz,256,0,stream>>>(A,W,bb,gg,be,out,M_store,K,N_true,out_ld,n_store); break;
        default: k_gemm_wmma<8,1><<<gsz,256,0,stream>>>(A,W,bb,gg,be,out,M_store,K,N_true,out_ld,n_store); break;
      }
    } else {
      switch (nt4) {
        case 1: k_gemm_wmma<1,0><<<gsz,256,0,stream>>>(A,W,bb,gg,be,out,M_store,K,N_true,out_ld,n_store); break;
        case 2: k_gemm_wmma<2,0><<<gsz,256,0,stream>>>(A,W,bb,gg,be,out,M_store,K,N_true,out_ld,n_store); break;
        case 4: k_gemm_wmma<4,0><<<gsz,256,0,stream>>>(A,W,bb,gg,be,out,M_store,K,N_true,out_ld,n_store); break;
        default: k_gemm_wmma<8,0><<<gsz,256,0,stream>>>(A,W,bb,gg,be,out,M_store,K,N_true,out_ld,n_store); break;
      }
    }
  };

  // ---- phase 1: scatter mean + gather by voxel type ----
  hipMemsetAsync(sums, 0, (size_t)NT_TYPES * 32 * 4, stream);
  hipMemsetAsync(cnt, 0, (size_t)NT_TYPES * 4, stream);
  k_scatter_local<<<(NL_ * 32 + 255) / 256, 256, 0, stream>>>(local_x, local_type, sums, cnt);
  k_gather_matched<<<(MP_ * 32 + 255) / 256, 256, 0, stream>>>(sums, cnt, voxel_type, matched);

  // ---- MFE: 32 -> 256 -> 256 -> 256 (LN + lrelu) ----
  gemm(matched, Wt[0], mfe_b[0], mfe_g[0], mfe_be[0], hb1, MP_, 32, 256, 1, 256, 256, 256);
  gemm(hb1,     Wt[1], mfe_b[1], mfe_g[1], mfe_be[1], hb2, MP_, 256, 256, 1, 256, 256, 256);
  gemm(hb2,     Wt[2], mfe_b[2], mfe_g[2], mfe_be[2], enc, MP_, 256, 256, 1, 256, 256, 256);

  // ---- MLP: concat(enc,voxel,z)=448 -> 512 -> 512 -> 512 ----
  k_pack448<<<dim3(2, MP_), 256, 0, stream>>>(enc, voxel_x, zf, hb1);
  gemm(hb1, Wt[3], mlp_b[0], mlp_g[0], mlp_be[0], hb2, MP_, 448, 512, 1, 512, 512, 512);
  gemm(hb2, Wt[4], mlp_b[1], mlp_g[1], mlp_be[1], hb1, MP_, 512, 512, 1, 512, 512, 512);
  gemm(hb1, Wt[5], mlp_b[2], mlp_g[2], mlp_be[2], xb,  MP_, 512, 512, 1, 512, 512, 512);

  // ---- degrees (once) ----
  const int* esrc = edge_index;
  const int* edst = edge_index + NE_;
  hipMemsetAsync(deg, 0, (size_t)MP_ * 4, stream);
  k_edge_deg<<<(NE_ + 255) / 256, 256, 0, stream>>>(edst, deg);
  k_dis<<<(MP_ + 255) / 256, 256, 0, stream>>>(deg, dis, dis2);

  // ---- GCN: 6 layers ----
  const int cK[6] = { 512, 256, 128, 64, 128, 256 };
  const int cN[6] = { 256, 128, 64, 128, 256, 512 };
  const half_t* ein = xb;
  half_t* eping[2] = { hb1, hb2 };
  for (int l = 0; l < 6; ++l) {
    int K = cK[l], N = cN[l];
    half_t* eout = eping[l & 1];
    gemm(ein, Wt[6 + l], nullptr, nullptr, nullptr, hbuff, MP_, K, N, 0, N, N, N);
    hipMemsetAsync(agg, 0, (size_t)NV_ * N * 4, stream);
    {
      int logq = 31 - __builtin_clz(N >> 2);      // N/4 is a power of two
      long long n = (long long)NE_ << logq;
      k_edge_agg<<<(unsigned)((n + 255) / 256), 256, 0, stream>>>(esrc, edst, dis, hbuff, agg, N, logq);
    }
    k_gcn_combine<<<dim3((N + 255) / 256, NV_), 256, 0, stream>>>(hbuff, agg, dis2, conv_b[l], N);
    hipMemsetAsync(fsum, 0, (size_t)N * 4, stream);
    k_colsum<<<(NV_ + 127) / 128, 256, 0, stream>>>(hbuff, fsum, N);
    k_meanfin<<<1, 512, 0, stream>>>(fsum, fmean, N);
    hipMemsetAsync(fsq, 0, (size_t)N * 4, stream);
    k_csqsum<<<(NV_ + 127) / 128, 256, 0, stream>>>(hbuff, gn_a[l], fmean, fsq, N);
    k_varfin<<<1, 512, 0, stream>>>(fsq, frstd, N);
    k_gnorm_out<<<dim3((N + 255) / 256, MP_), 256, 0, stream>>>(hbuff, gn_a[l], fmean, frstd,
                                                                gn_w[l], gn_b[l], eout, N);
    ein = eout;
  }
  const half_t* efin = ein;   // hb2 after 6 layers

  // ---- decoder: concat(e,x,enc,voxel,z)=1472 -> 512 -> 256 -> 128 -> 64 -> 8 ----
  k_pack1472<<<dim3(6, MP_), 256, 0, stream>>>(efin, xb, enc, voxel_x, zf, concat);
  gemm(concat, Wt[12], dec_b[0], dec_g[0], dec_be[0], hb1, MP_, 1472, 512, 1, 512, 512, 512);
  gemm(hb1,    Wt[13], dec_b[1], dec_g[1], dec_be[1], xb,  MP_, 512, 256, 1, 256, 256, 256);
  gemm(xb,     Wt[14], dec_b[2], dec_g[2], dec_be[2], hb1, MP_, 256, 128, 1, 128, 128, 128);
  gemm(hb1,    Wt[15], dec_b[3], dec_g[3], dec_be[3], xb,  MP_, 128, 64, 1, 64, 64, 64);

  float* outp = (float*)d_out;                       // logits [NV_,8]
  gemm(xb, Wt[16], dec_b[4], nullptr, nullptr, outp, NV_, 64, 64, 0, 8, 8, 8);

  // ---- gumbel softmax outputs ----
  k_gumbel<<<(NV_ + 255) / 256, 256, 0, stream>>>(outp, gumbel,
                                                  outp + (size_t)NV_ * 8,    // label_hard
                                                  outp + (size_t)NV_ * 16);  // label_soft
}
